// GCNConv_73083163509339
// MI455X (gfx1250) — matrix operand order repaired and checked by reading, not compiled, not run on hardware
//
#include <hip/hip_runtime.h>
#include <math.h>

// ---------------------------------------------------------------------------
// GCN block on MI455X (gfx1250, wave32, WMMA 16x16x32 bf16, TDM staging).
//   1. pack weights fp32 -> bf16 in WMMA-B fragment order (ws)
//   2. adjacency: zero -> scatter 1.0 -> zero diag (bf16, exact 0/1, 33.5MB)
//   3. pack x -> bf16 pairs into cat[:, 0:128]
//   4. lin0: h = x@W0+b0 via WMMA -> H2 in K-pair-interleaved layout
//      (rows k/2, cols (n,parity)) so agg B-fragments are dword loads
//   5. agg = adj @ H (34.4 GFLOP bf16 WMMA); A/B tiles staged to LDS by the
//      Tensor Data Mover, double-buffered, s_wait_tensorcnt synchronized
//   6. fused: cat@W1+b1 -> @W2+b2 -> exact GELU -> @Wo+bo -> x + out (fp32)
// ---------------------------------------------------------------------------

typedef __bf16 bf16;
typedef __attribute__((ext_vector_type(16))) __bf16        v16bf;
typedef __attribute__((ext_vector_type(8)))  float         v8f;
typedef __attribute__((ext_vector_type(8)))  unsigned int  v8u;
typedef __attribute__((ext_vector_type(4)))  unsigned int  v4u;
typedef __attribute__((ext_vector_type(8)))  int           v8i_t;
typedef __attribute__((ext_vector_type(4)))  int           v4i_t;

static constexpr int Bc = 8;       // batch
static constexpr int Nn = 4096;    // nodes
static constexpr int Cc = 128;     // channels
static constexpr int NB = Bc * Cc; // 1024 cols of H/agg

#if defined(__gfx1250__) && __has_builtin(__builtin_amdgcn_tensor_load_to_lds) && \
    __has_builtin(__builtin_amdgcn_s_wait_tensorcnt)
#define HAVE_TDM 1
#else
#define HAVE_TDM 0
#endif

struct U32x8 { uint4 a, b; };

__device__ __forceinline__ bf16 f2bf(float f) { return (bf16)f; }

__device__ __forceinline__ unsigned pack2bf(float a, float b) {
  unsigned short ua = __builtin_bit_cast(unsigned short, (bf16)a);
  unsigned short ub = __builtin_bit_cast(unsigned short, (bf16)b);
  return (unsigned)ua | ((unsigned)ub << 16);
}

__device__ __forceinline__ v8f wmma_bf(v16bf a, v16bf b, v8f c) {
  // (neg_a, A, neg_b, B, c_mod, C, reuse_a, reuse_b)
  return __builtin_amdgcn_wmma_f32_16x16x32_bf16(false, a, false, b, (short)0, c,
                                                 false, false);
}

// A fragment (16x32 bf16) from row-major bf16 src, ld elements (ld%8==0).
// Per lane: elements 0..7 = K[hi*8 .. hi*8+7], 8..15 = K[16+hi*8 ..] -> two
// contiguous 16B loads, zero shuffle VALU.
__device__ __forceinline__ v16bf load_a_frag_b128(const bf16* src, int ld) {
  int lane = threadIdx.x & 31;
  const bf16* p = src + (lane & 15) * ld + (lane >> 4) * 8;
  U32x8 t;
  t.a = *(const uint4*)(p);
  t.b = *(const uint4*)(p + 16);
  return __builtin_bit_cast(v16bf, t);
}

// B fragment from fragment-order packed weights: 8 consecutive dwords / lane.
__device__ __forceinline__ v16bf load_b_frag_packed(const unsigned* P, int kt, int ct) {
  int lane = threadIdx.x & 31;
  const unsigned* p = P + ((size_t)((kt * 8 + ct) * 32 + lane) * 8);
  U32x8 t;
  t.a = *(const uint4*)(p);
  t.b = *(const uint4*)(p + 4);
  return __builtin_bit_cast(v16bf, t);
}

// B fragment from LDS tile of pair-interleaved H2: rows = k-pair, 264 bf16
// stride; per lane 8 aligned dword loads (pair (k,k+1) per dword).
__device__ __forceinline__ v16bf load_b_frag_pairs(const bf16* Bst, int nloc) {
  int hi = (threadIdx.x & 31) >> 4;
  v8u b;
#pragma unroll
  for (int j = 0; j < 8; ++j)
    b[j] = *(const unsigned*)(Bst + (size_t)((j >> 2) * 8 + hi * 4 + (j & 3)) * 264 + nloc * 2);
  return __builtin_bit_cast(v16bf, b);
}

#if HAVE_TDM
// Issue a 2D tensor_load_to_lds: 2-byte elements, LDS padding per descriptor.
__device__ __forceinline__ void tdm_load_2d(unsigned lds_off, unsigned long long ga,
                                            unsigned g1w0, unsigned td0, unsigned td1,
                                            unsigned tile0, unsigned tile1,
                                            unsigned stride0) {
  v4u g0 = { 1u,                                      // count=1, user desc
             lds_off,                                  // lds_addr (bytes)
             (unsigned)ga,                             // global_addr lo
             (unsigned)((ga >> 32) & 0x01FFFFFFu) | (2u << 30) };  // hi | type=2
  v8i_t g1 = { (int)g1w0,                              // data_size/pad config
               (int)((td0 & 0xFFFFu) << 16),           // tensor_dim0 lo16
               (int)((td0 >> 16) | ((td1 & 0xFFFFu) << 16)),
               (int)((td1 >> 16) | (tile0 << 16)),
               (int)tile1,                             // tile_dim1 (tile_dim2=0)
               (int)stride0,                           // tensor_dim0_stride lo32
               0, 0 };
  v4i_t z = { 0, 0, 0, 0 };
#if __clang_major__ >= 23
  v8i_t z8 = { 0, 0, 0, 0, 0, 0, 0, 0 };
  __builtin_amdgcn_tensor_load_to_lds(g0, g1, z, z, z8, 0);
#else
  __builtin_amdgcn_tensor_load_to_lds(g0, g1, z, z, 0);
#endif
}
// data_size=2B (code 1), pad_enable, pad_interval code, pad_amount code 3 (4 DW)
#define G1W0_A ((1u << 16) | (1u << 20) | (3u << 22) | (3u << 25))  // 16DW rows -> stride 40 bf16
#define G1W0_B ((1u << 16) | (1u << 20) | (6u << 22) | (3u << 25))  // 128DW rows -> stride 264 bf16
#endif

// ---------------------------------------------------------------------------
__device__ __forceinline__ void pack_w_frag(const float* W, unsigned* P, int idx) {
  // idx (dwords): ((kt*8 + ct)*32 + lane)*8 + j ; pair (k, k+1) x col n
  int j = idx & 7, lane = (idx >> 3) & 31, rem = idx >> 8;
  int ct = rem & 7, kt = rem >> 3;
  int k = kt * 32 + (j >> 2) * 16 + (lane >> 4) * 8 + 2 * (j & 3);
  int n = ct * 16 + (lane & 15);
  P[idx] = pack2bf(W[k * 128 + n], W[(k + 1) * 128 + n]);
}

__global__ void k_pack_weights(const float* W0, const float* W1,
                               const float* W2, const float* Wo,
                               unsigned* W0p, unsigned* W1p,
                               unsigned* W2p, unsigned* Wop) {
  int idx = blockIdx.x * blockDim.x + threadIdx.x;   // 0..16383
  pack_w_frag(W1, W1p, idx);                         // K=256: 16384 dwords
  if (idx < 8192) {                                  // K=128 matrices
    pack_w_frag(W0, W0p, idx);
    pack_w_frag(W2, W2p, idx);
    pack_w_frag(Wo, Wop, idx);
  }
}

__global__ void k_zero16(uint4* p, size_t n16) {
  size_t i = (size_t)blockIdx.x * blockDim.x + threadIdx.x;
  if (i < n16) { uint4 z; z.x = z.y = z.z = z.w = 0u; p[i] = z; }
}

__global__ void k_adj_scatter(const int* ei, int E, bf16* adj) {
  int e = blockIdx.x * blockDim.x + threadIdx.x;
  if (e < E) {
    int u = ei[e], v = ei[E + e];
    adj[(size_t)u * Nn + v] = f2bf(1.0f);   // all writers store 1.0 -> race-safe,
    adj[(size_t)v * Nn + u] = f2bf(1.0f);   // duplicates collapse like .set
  }
}

__global__ void k_adj_diag(bf16* adj) {
  int i = blockIdx.x * blockDim.x + threadIdx.x;
  if (i < Nn) adj[(size_t)i * Nn + i] = f2bf(0.0f);
}

__global__ void k_pack_x(const float* x, unsigned* catu) {
  size_t idx = (size_t)blockIdx.x * blockDim.x + threadIdx.x;  // pair index
  if (idx < (size_t)Bc * Nn * 64) {
    size_t row = idx >> 6;
    int cp = (int)(idx & 63);
    catu[row * 128 + cp] = pack2bf(x[row * 128 + 2 * cp], x[row * 128 + 2 * cp + 1]);
  }
}

// lin0: one wave per 16-node tile of one batch; writes H2 pair-interleaved:
// dword at [(k>>1)*1024 + col] = bf16 pair (h[k], h[k+1]).
__global__ __launch_bounds__(128) void k_lin0(const float* __restrict__ x,
                                              const unsigned* __restrict__ W0p,
                                              const float* __restrict__ b0,
                                              unsigned* __restrict__ H2u) {
  int wave = threadIdx.x >> 5, lane = threadIdx.x & 31;
  int lo = lane & 15, hi = lane >> 4;
  int b = blockIdx.x >> 6;
  int i0 = ((blockIdx.x & 63) * 4 + wave) * 16;

  // A fragments: fp32 rows -> packed bf16 pairs (v_cvt_pk_bf16_f32)
  v16bf a[4];
  const float* xrow = x + (size_t)(b * Nn + i0 + lo) * Cc;
#pragma unroll
  for (int kt = 0; kt < 4; ++kt) {
    v8u au;
#pragma unroll
    for (int i = 0; i < 8; ++i) {
      int k = kt * 32 + ((i & 3) * 2) + ((i >> 2) * 16) + hi * 8;
      au[i] = pack2bf(xrow[k], xrow[k + 1]);
    }
    a[kt] = __builtin_bit_cast(v16bf, au);
  }
#pragma unroll
  for (int ct = 0; ct < 8; ++ct) {
    v8f acc = {};
#pragma unroll
    for (int kt = 0; kt < 4; ++kt)
      acc = wmma_bf(a[kt], load_b_frag_packed(W0p, kt, ct), acc);
    int c = ct * 16 + lo;
    float bias = b0[c];
#pragma unroll
    for (int t = 0; t < 4; ++t) {                 // rows hi*8+2t, +1 are in-lane
      int m0 = hi * 8 + 2 * t;
      H2u[(size_t)((i0 + m0) >> 1) * NB + b * Cc + c] =
          pack2bf(acc[2 * t] + bias, acc[2 * t + 1] + bias);
    }
  }
}

// agg = adj(4096x4096) @ H(4096x1024). Workgroup tile 64x128, 8 waves (4x2).
// TDM double-buffered LDS staging; wave16x64 via 4 accumulators.
__global__ __launch_bounds__(256) void k_agg(const bf16* __restrict__ adj,
                                             const bf16* __restrict__ H2,
                                             bf16* __restrict__ cat) {
  __shared__ alignas(16) bf16 As[2][64][40];    // 32 bf16 rows + TDM pad
  __shared__ alignas(16) bf16 Bs[2][16][264];   // 256 bf16 rows + TDM pad
  const int i0 = blockIdx.x * 64;
  const int n0 = blockIdx.y * 128;
  const int tid = threadIdx.x;
  const int wave = tid >> 5, lane = tid & 31, lo = lane & 15, hi = lane >> 4;
  const int mrow = (wave & 3) * 16;
  const int ncol = (wave >> 2) * 64;

  v8f zero = {};
  v8f acc[4];
#pragma unroll
  for (int t = 0; t < 4; ++t) acc[t] = zero;

#if HAVE_TDM
  const unsigned ldsA[2] = { (unsigned)(size_t)&As[0][0][0],
                             (unsigned)(size_t)&As[1][0][0] };
  const unsigned ldsB[2] = { (unsigned)(size_t)&Bs[0][0][0],
                             (unsigned)(size_t)&Bs[1][0][0] };
  const unsigned long long gaA = (unsigned long long)(size_t)adj +
                                 (unsigned long long)i0 * Nn * 2;
  const unsigned long long gaB = (unsigned long long)(size_t)H2 +
                                 (unsigned long long)n0 * 2 * 2;
  constexpr int T = Nn / 32;
  if (wave == 0) {      // prologue: tile 0 into buffer 0 (EXEC-independent DMA)
    tdm_load_2d(ldsA[0], gaA, G1W0_A, Nn, Nn, 32, 64, Nn);
    tdm_load_2d(ldsB[0], gaB, G1W0_B, 2 * NB, Nn / 2, 256, 16, 2 * NB);
  }
  for (int t = 0; t < T; ++t) {
    int cur = t & 1;
    if (wave == 0) {
      if (t + 1 < T) {
        int k0 = (t + 1) * 32;
        tdm_load_2d(ldsA[1 - cur], gaA + (unsigned long long)k0 * 2,
                    G1W0_A, Nn, Nn, 32, 64, Nn);
        tdm_load_2d(ldsB[1 - cur], gaB + (unsigned long long)(k0 >> 1) * (2 * NB) * 2,
                    G1W0_B, 2 * NB, Nn / 2, 256, 16, 2 * NB);
        __builtin_amdgcn_s_wait_tensorcnt(2);   // tile t's pair complete
      } else {
        __builtin_amdgcn_s_wait_tensorcnt(0);
      }
    }
    __syncthreads();                            // publish tile t to all waves
    v16bf af = load_a_frag_b128(&As[cur][mrow][0], 40);
#pragma unroll
    for (int ct = 0; ct < 4; ++ct)
      acc[ct] = wmma_bf(af, load_b_frag_pairs(&Bs[cur][0][0], ncol + ct * 16 + lo),
                        acc[ct]);
    __syncthreads();                            // buffer reusable for tile t+2
  }
#else
  // Fallback: single-buffer manual staging (no TDM builtin available).
  int ar = tid >> 2, ac = (tid & 3) * 8;
  for (int k0 = 0; k0 < Nn; k0 += 32) {
    *(uint4*)&As[0][ar][ac] = *(const uint4*)&adj[(size_t)(i0 + ar) * Nn + k0 + ac];
#pragma unroll
    for (int t = 0; t < 2; ++t) {
      int ch = tid + t * 256;                    // 512 chunks of 8 bf16
      int br = ch >> 5, bc = (ch & 31) * 8;
      *(uint4*)&Bs[0][br][bc] =
          *(const uint4*)&H2[(size_t)((k0 >> 1) + br) * (2 * NB) + n0 * 2 + bc];
    }
    __syncthreads();
    if (k0 + 32 < Nn)
      __builtin_prefetch(&adj[(size_t)(i0 + ar) * Nn + k0 + 32 + ac], 0, 1);
    v16bf af = load_a_frag_b128(&As[0][mrow][0], 40);
#pragma unroll
    for (int ct = 0; ct < 4; ++ct)
      acc[ct] = wmma_bf(af, load_b_frag_pairs(&Bs[0][0][0], ncol + ct * 16 + lo),
                        acc[ct]);
    __syncthreads();
  }
#endif

#pragma unroll
  for (int ct = 0; ct < 4; ++ct) {
    int col = n0 + ncol + ct * 16 + lo;          // 0..1023 -> (b, c)
    int bb = col >> 7, c = col & 127;
#pragma unroll
    for (int r = 0; r < 8; ++r) {
      int i = i0 + mrow + r + hi * 8;
      cat[((size_t)bb * Nn + i) * 256 + 128 + c] = f2bf(acc[ct][r]);
    }
  }
}

// Fused MLP: per wave one 16-row tile of cat; three WMMA GEMMs through LDS,
// exact GELU, fp32 residual output.
__global__ __launch_bounds__(128) void k_mlp(const float* __restrict__ x,
                                             const bf16* __restrict__ cat,
                                             const unsigned* __restrict__ W1p,
                                             const float* __restrict__ b1,
                                             const unsigned* __restrict__ W2p,
                                             const float* __restrict__ b2,
                                             const unsigned* __restrict__ Wop,
                                             const float* __restrict__ bo,
                                             float* __restrict__ out) {
  __shared__ alignas(16) bf16 T1[4][16][136];
  __shared__ alignas(16) bf16 G[4][16][136];
  int wave = threadIdx.x >> 5, lane = threadIdx.x & 31;
  int lo = lane & 15, hi = lane >> 4;
  size_t row0 = ((size_t)blockIdx.x * 4 + wave) * 16;

  // --- GEMM1: cat(16x256) @ W1(256x128) ---
  v16bf a1[8];
  const bf16* cbase = cat + row0 * 256;
#pragma unroll
  for (int kt = 0; kt < 8; ++kt) a1[kt] = load_a_frag_b128(cbase + kt * 32, 256);
#pragma unroll
  for (int ct = 0; ct < 8; ++ct) {
    v8f acc = {};
#pragma unroll
    for (int kt = 0; kt < 8; ++kt)
      acc = wmma_bf(a1[kt], load_b_frag_packed(W1p, kt, ct), acc);
    int c = ct * 16 + lo;
    float bias = b1[c];
#pragma unroll
    for (int r = 0; r < 8; ++r) T1[wave][r + hi * 8][c] = f2bf(acc[r] + bias);
  }

  // --- GEMM2 + exact GELU: T1(16x128) @ W2(128x128) ---
  v16bf a2[4];
#pragma unroll
  for (int kt = 0; kt < 4; ++kt) a2[kt] = load_a_frag_b128(&T1[wave][0][kt * 32], 136);
#pragma unroll
  for (int ct = 0; ct < 8; ++ct) {
    v8f acc = {};
#pragma unroll
    for (int kt = 0; kt < 4; ++kt)
      acc = wmma_bf(a2[kt], load_b_frag_packed(W2p, kt, ct), acc);
    int c = ct * 16 + lo;
    float bias = b2[c];
#pragma unroll
    for (int r = 0; r < 8; ++r) {
      float v = acc[r] + bias;
      v = 0.5f * v * (1.0f + erff(v * 0.7071067811865476f));  // torch-exact GELU
      G[wave][r + hi * 8][c] = f2bf(v);
    }
  }

  // --- GEMM3 + residual: G(16x128) @ Wo(128x128) ---
  v16bf a3[4];
#pragma unroll
  for (int kt = 0; kt < 4; ++kt) a3[kt] = load_a_frag_b128(&G[wave][0][kt * 32], 136);
#pragma unroll
  for (int ct = 0; ct < 8; ++ct) {
    v8f acc = {};
#pragma unroll
    for (int kt = 0; kt < 4; ++kt)
      acc = wmma_bf(a3[kt], load_b_frag_packed(Wop, kt, ct), acc);
    int c = ct * 16 + lo;
    float bias = bo[c];
#pragma unroll
    for (int r = 0; r < 8; ++r) {
      size_t row = row0 + r + hi * 8;
      out[row * 128 + c] = x[row * 128 + c] + acc[r] + bias;
    }
  }
}

// ---------------------------------------------------------------------------
extern "C" void kernel_launch(void* const* d_in, const int* in_sizes, int n_in,
                              void* d_out, int out_size, void* d_ws, size_t ws_size,
                              hipStream_t stream) {
  const float* x  = (const float*)d_in[0];
  const int*   ei = (const int*)d_in[1];
  const float* W0 = (const float*)d_in[2];
  const float* b0 = (const float*)d_in[3];
  const float* W1 = (const float*)d_in[4];
  const float* b1 = (const float*)d_in[5];
  const float* W2 = (const float*)d_in[6];
  const float* b2 = (const float*)d_in[7];
  const float* Wo = (const float*)d_in[8];
  const float* bo = (const float*)d_in[9];
  float* out = (float*)d_out;
  int E = in_sizes[1] / 2;

  // workspace layout (bytes)
  char* ws = (char*)d_ws;
  bf16*     adj = (bf16*)(ws);                       // 4096*4096*2 = 33,554,432
  bf16*     H2  = (bf16*)(ws + 33554432);            // 4096*1024*2 =  8,388,608
  bf16*     cat = (bf16*)(ws + 41943040);            // 32768*256*2 = 16,777,216
  unsigned* W0p = (unsigned*)(ws + 58720256);        // 32 KB
  unsigned* W1p = (unsigned*)(ws + 58720256 + 32768);   // 64 KB
  unsigned* W2p = (unsigned*)(ws + 58720256 + 98304);   // 32 KB
  unsigned* Wop = (unsigned*)(ws + 58720256 + 131072);  // 32 KB (end ~56.2 MB)

  k_pack_weights<<<64, 256, 0, stream>>>(W0, W1, W2, Wo, W0p, W1p, W2p, Wop);
  k_zero16<<<8192, 256, 0, stream>>>((uint4*)adj, (size_t)Nn * Nn * 2 / 16);
  k_adj_scatter<<<(E + 255) / 256, 256, 0, stream>>>(ei, E, adj);
  k_adj_diag<<<16, 256, 0, stream>>>(adj);
  k_pack_x<<<8192, 256, 0, stream>>>(x, (unsigned*)cat);
  k_lin0<<<512, 128, 0, stream>>>(x, W0p, b0, (unsigned*)H2);
  k_agg<<<dim3(64, 8), 256, 0, stream>>>(adj, H2, cat);
  k_mlp<<<512, 128, 0, stream>>>(x, cat, W1p, b1, W2p, b2, Wop, bo, out);
}
